// Conv_48679159332865
// MI455X (gfx1250) — matrix operand order repaired and checked by reading, not compiled, not run on hardware
//
#include <hip/hip_runtime.h>
#include <hip/hip_bf16.h>
#include <stdint.h>

typedef unsigned short u16;
typedef __attribute__((ext_vector_type(16))) __bf16 v16bf;
typedef __attribute__((ext_vector_type(8)))  float  v8f;
typedef __attribute__((ext_vector_type(4)))  float  v4f;
typedef __attribute__((ext_vector_type(8)))  u16    v8u;
typedef __attribute__((ext_vector_type(16))) u16    v16u;

#define V_CNT   4096
#define S_CNT   216
#define FIN_C   32
#define FOUT_C  32
#define K_C     8
#define NNZ_C   36864
#define D_C     (S_CNT * FIN_C)            /* 6912 columns of the SpMM operand   */
#define NROWS   (V_CNT * S_CNT)            /* 884736 GEMM rows                   */
#define SLABE   ((size_t)NROWS * FIN_C)    /* elements (bf16) per Chebyshev slab */

__device__ __forceinline__ float bf2f(u16 b) {
    union { unsigned int u; float f; } x; x.u = ((unsigned int)b) << 16; return x.f;
}
__device__ __forceinline__ u16 f2bf(float f) {
    union { float f; unsigned int u; } x; x.f = f;
    unsigned int r = x.u + 0x7FFFu + ((x.u >> 16) & 1u);   // round-to-nearest-even
    return (u16)(r >> 16);
}

// ---------------------------------------------------------------- CSR build --
__global__ void k_zero_i32(int* __restrict__ p, int n) {
    int i = blockIdx.x * blockDim.x + threadIdx.x;
    if (i < n) p[i] = 0;
}

__global__ void k_row_hist(const int* __restrict__ rows, int* __restrict__ cnt) {
    int e = blockIdx.x * blockDim.x + threadIdx.x;
    if (e < NNZ_C) atomicAdd(&cnt[rows[e]], 1);
}

// single-block exclusive scan over 4096 counts -> row_start[4097] and cursor copy
__global__ __launch_bounds__(1024) void k_scan4096(const int* __restrict__ cnt,
                                                   int* __restrict__ rstart,
                                                   int* __restrict__ cursor) {
    __shared__ int sd[1024];
    int t = threadIdx.x;
    int v0 = cnt[4*t], v1 = cnt[4*t+1], v2 = cnt[4*t+2], v3 = cnt[4*t+3];
    int sum = v0 + v1 + v2 + v3;
    sd[t] = sum; __syncthreads();
    for (int off = 1; off < 1024; off <<= 1) {
        int x = (t >= off) ? sd[t - off] : 0;
        __syncthreads();
        sd[t] += x;
        __syncthreads();
    }
    int incl = sd[t], excl = incl - sum;
    int p0 = excl, p1 = excl + v0, p2 = p1 + v1, p3 = p2 + v2;
    rstart[4*t] = p0; rstart[4*t+1] = p1; rstart[4*t+2] = p2; rstart[4*t+3] = p3;
    cursor[4*t] = p0; cursor[4*t+1] = p1; cursor[4*t+2] = p2; cursor[4*t+3] = p3;
    if (t == 1023) rstart[4096] = incl;   // == NNZ
}

__global__ void k_csr_scatter(const int* __restrict__ rows, const int* __restrict__ cols,
                              const float* __restrict__ vals, int* __restrict__ cursor,
                              int* __restrict__ ccol, float* __restrict__ cval) {
    int e = blockIdx.x * blockDim.x + threadIdx.x;
    if (e >= NNZ_C) return;
    int r = rows[e];
    int pos = atomicAdd(&cursor[r], 1);
    ccol[pos] = cols[e];
    cval[pos] = vals[e];
}

// ------------------------------------------------------------- data packing --
// weight [K][FIN][FOUT] fp32 -> Wt[fout][k*32+fin] bf16 (B-fragment friendly)
__global__ void k_pack_w(const float* __restrict__ w, u16* __restrict__ wt) {
    int i = blockIdx.x * blockDim.x + threadIdx.x;
    if (i >= K_C * FIN_C * FOUT_C) return;
    int kfin = i >> 5, fout = i & 31;
    wt[fout * 256 + kfin] = f2bf(w[i]);
}

// inputs [FIN][V][S] fp32 -> T0 [v][s][fin] bf16
__global__ void k_pack_x0(const float* __restrict__ in, u16* __restrict__ t0) {
    int i = blockIdx.x * blockDim.x + threadIdx.x;   // total = NROWS*32, exact grid
    int fin  = i & 31;
    int rest = i >> 5;                               // v*216 + s
    t0[i] = f2bf(in[(size_t)fin * NROWS + rest]);
}

// ----------------------------------------------- fused SpMM + Cheb combine ---
// T_out[r,c] = alpha * sum_j cval[j]*T_prev[ccol[j],c]  + beta * T_pp[r,c]
// one workgroup per row r; 256 threads sweep 6912 columns (coalesced gathers)
__global__ __launch_bounds__(256) void k_spmm_cheb(
    const u16* __restrict__ Tprev, const u16* __restrict__ Tpp, u16* __restrict__ Tout,
    const int* __restrict__ rstart, const int* __restrict__ ccol,
    const float* __restrict__ cval, float alpha, float beta) {
    int r = blockIdx.x;
    int start = rstart[r], end = rstart[r + 1];      // uniform -> SALU/SMEM
    size_t rbase = (size_t)r * D_C;
    for (int c = threadIdx.x; c < D_C; c += 256) {
        float acc = 0.f;
        for (int j = start; j < end; ++j) {
            int col = ccol[j];
            acc = fmaf(cval[j], bf2f(Tprev[(size_t)col * D_C + c]), acc);
        }
        float res = alpha * acc;
        if (beta != 0.f) res = fmaf(beta, bf2f(Tpp[rbase + c]), res);
        Tout[rbase + c] = f2bf(res);
    }
}

// ----------------------------------------------------------- WMMA projection --
// out[fout, n] = bias[fout] + sum_{k,fin} T_k[n,fin] * W[k,fin,fout]
// 8 waves/block, wave = 16 rows x 32 fout (two 16x16 f32 tiles), 8 K-chunks of 32
__global__ __launch_bounds__(256) void k_cheb_gemm(
    const u16* __restrict__ Tbase,   // 8 contiguous slabs, stride SLABE
    const u16* __restrict__ Wt,      // [32][256] bf16
    const float* __restrict__ bias,
    float* __restrict__ out) {
    __shared__ u16 wlds[FOUT_C * 256];               // 16 KB
    for (int i = threadIdx.x; i < (FOUT_C * 256) / 8; i += 256)
        ((v8u*)wlds)[i] = ((const v8u*)Wt)[i];
    __syncthreads();

    int wave = threadIdx.x >> 5;                     // wave32
    int lane = threadIdx.x & 31;
    int n0   = blockIdx.x * 128 + wave * 16;
    int row  = n0 + (lane & 15);                     // A: M = lane%16
    int kbA  = (lane >> 4) * 16;                     // A: lanes<16 -> K0..15
    int col  = lane & 15;                            // B/D: N = lane%16
    int kbB  = (lane >> 4) * 16;                     // B: lanes<16 -> K0..15

    v8f acc0 = {0,0,0,0,0,0,0,0};
    v8f acc1 = {0,0,0,0,0,0,0,0};

    #pragma unroll
    for (int kk = 0; kk < K_C; ++kk) {
        const u16* ap = Tbase + (size_t)kk * SLABE + (size_t)row * 32 + kbA;
        __builtin_prefetch(ap + SLABE, 0, 0);        // global_prefetch next slab
        union { v16u v; v8u h[2]; } a;
        a.h[0] = *(const v8u*)(ap);                  // K base+0 .. +7
        a.h[1] = *(const v8u*)(ap + 8);              // K base+8 .. +15
        v16bf af = __builtin_bit_cast(v16bf, a.v);

        const u16* bp0 = wlds + col * 256 + kk * 32 + kbB;
        union { v16u v; v8u h[2]; } b0, b1;
        b0.h[0] = *(const v8u*)(bp0);
        b0.h[1] = *(const v8u*)(bp0 + 8);
        const u16* bp1 = bp0 + 16 * 256;
        b1.h[0] = *(const v8u*)(bp1);
        b1.h[1] = *(const v8u*)(bp1 + 8);
        v16bf bf0 = __builtin_bit_cast(v16bf, b0.v);
        v16bf bf1 = __builtin_bit_cast(v16bf, b1.v);

        acc0 = __builtin_amdgcn_wmma_f32_16x16x32_bf16(false, af, false, bf0,
                                                       (short)0, acc0, false, false);
        acc1 = __builtin_amdgcn_wmma_f32_16x16x32_bf16(false, af, false, bf1,
                                                       (short)0, acc1, false, false);
    }

    float b0 = bias[col], b1 = bias[col + 16];
    size_t nr = (size_t)n0 + (size_t)(lane >> 4) * 8;          // 8 consecutive rows
    float* o0 = out + (size_t)col * NROWS + nr;
    float* o1 = out + (size_t)(col + 16) * NROWS + nr;
    v4f s;
    s[0]=acc0[0]+b0; s[1]=acc0[1]+b0; s[2]=acc0[2]+b0; s[3]=acc0[3]+b0;
    *(v4f*)(o0)     = s;
    s[0]=acc0[4]+b0; s[1]=acc0[5]+b0; s[2]=acc0[6]+b0; s[3]=acc0[7]+b0;
    *(v4f*)(o0 + 4) = s;
    s[0]=acc1[0]+b1; s[1]=acc1[1]+b1; s[2]=acc1[2]+b1; s[3]=acc1[3]+b1;
    *(v4f*)(o1)     = s;
    s[0]=acc1[4]+b1; s[1]=acc1[5]+b1; s[2]=acc1[6]+b1; s[3]=acc1[7]+b1;
    *(v4f*)(o1 + 4) = s;
}

// ----------------------------------------------------------------- launcher --
extern "C" void kernel_launch(void* const* d_in, const int* in_sizes, int n_in,
                              void* d_out, int out_size, void* d_ws, size_t ws_size,
                              hipStream_t stream) {
    const float* inputs  = (const float*)d_in[0];
    const float* weight  = (const float*)d_in[1];
    const float* bias    = (const float*)d_in[2];
    const float* lapv    = (const float*)d_in[3];
    const int*   laprows = (const int*)d_in[4];
    const int*   lapcols = (const int*)d_in[5];
    float* out = (float*)d_out;

    // workspace carve-up (256B aligned)
    char* ws = (char*)d_ws;
    size_t off = 0;
    auto take = [&](size_t bytes) { char* p = ws + off;
                                    off = (off + bytes + 255) & ~(size_t)255; return p; };
    u16*  tslabs  = (u16*)take(8 * SLABE * sizeof(u16));     // 453 MB: T0..T7 bf16
    u16*  wt      = (u16*)take(FOUT_C * 256 * sizeof(u16));
    int*  cnt     = (int*)take(V_CNT * sizeof(int));
    int*  rstart  = (int*)take((V_CNT + 1) * sizeof(int));
    int*  cursor  = (int*)take(V_CNT * sizeof(int));
    int*  ccol    = (int*)take(NNZ_C * sizeof(int));
    float* cval   = (float*)take(NNZ_C * sizeof(float));
    (void)ws_size; (void)in_sizes; (void)n_in; (void)out_size;

    // 1) device CSR build (counting sort; NNZ is tiny)
    k_zero_i32<<<(V_CNT + 255) / 256, 256, 0, stream>>>(cnt, V_CNT);
    k_row_hist<<<(NNZ_C + 255) / 256, 256, 0, stream>>>(laprows, cnt);
    k_scan4096<<<1, 1024, 0, stream>>>(cnt, rstart, cursor);
    k_csr_scatter<<<(NNZ_C + 255) / 256, 256, 0, stream>>>(laprows, lapcols, lapv,
                                                           cursor, ccol, cval);
    // 2) pack weight + T0 into bf16 slabs
    k_pack_w<<<(K_C * FIN_C * FOUT_C + 255) / 256, 256, 0, stream>>>(weight, wt);
    k_pack_x0<<<(NROWS * FIN_C) / 256, 256, 0, stream>>>(inputs, tslabs);

    // 3) Chebyshev recurrence: T1 = L*T0 ; Tk = 2*L*T(k-1) - T(k-2)
    u16* T[K_C];
    for (int k = 0; k < K_C; ++k) T[k] = tslabs + (size_t)k * SLABE;
    k_spmm_cheb<<<V_CNT, 256, 0, stream>>>(T[0], nullptr, T[1], rstart, ccol, cval,
                                           1.0f, 0.0f);
    for (int k = 2; k < K_C; ++k)
        k_spmm_cheb<<<V_CNT, 256, 0, stream>>>(T[k-1], T[k-2], T[k], rstart, ccol, cval,
                                               2.0f, -1.0f);

    // 4) WMMA projection GEMM + bias, writes [FOUT, V*S] directly
    k_cheb_gemm<<<NROWS / 128, 256, 0, stream>>>(tslabs, wt, bias, out);
}
